// EMLLocalMessageBlock_3332894621837
// MI455X (gfx1250) — matrix operand order repaired and checked by reading, not compiled, not run on hardware
//
#include <hip/hip_runtime.h>
#include <hip/hip_bf16.h>
#include <math.h>

// ---------------------------------------------------------------------------
// EML local message block for MI455X (gfx1250, wave32, WMMA f16 16x16x32).
//
// h1(l,n) = center@Wc + Z(l,n)@Wn + relbias[n], where Z is the fixed permuted
// neighborhood gather implied by the torch reshape; values = Z@vW + vb.
// Wc = W1[0:128]+W1[256:384], Wn = W1[128:256]-W1[256:384],
// relbias[n] = rel_pos[n]@W1[384:392] + b1.
// ---------------------------------------------------------------------------

typedef _Float16 v16h __attribute__((ext_vector_type(16)));
typedef _Float16 h8   __attribute__((ext_vector_type(8)));
typedef _Float16 h4   __attribute__((ext_vector_type(4)));
typedef float    v8f  __attribute__((ext_vector_type(8)));

#define BATCH 2
#define HH 128
#define WW 128
#define HWSZ (HH*WW)
#define BL (BATCH*HWSZ)     // 32768 tokens
#define GEPS 1e-6f
#define LNE  1e-5f
#define CLIPV 3.0f
#define TPB 14              // tokens per mega-block: 14*9 = 126 rows -> 8 M-tiles

__device__ __forceinline__ v8f wmma_f16(v16h a, v16h b, v8f c) {
  return __builtin_amdgcn_wmma_f32_16x16x32_f16(
      false, a, false, b, (short)0, c, false, false);
}

// A-fragment per ISA layout: lane<16 -> row = lane, K = {k0..k0+7, k0+16..k0+23}
//                            lane>=16 -> row = lane-16, K = {k0+8..15, k0+24..31}
__device__ __forceinline__ v16h load_afrag(const _Float16* A, int row,
                                           int rowstride, int kc, int lane) {
  int koff = kc * 32 + ((lane >> 4) << 3);
  const _Float16* base = A + (size_t)row * rowstride + koff;
  h8 lo = *(const h8*)(base);
  h8 hi = *(const h8*)(base + 16);
  return __builtin_shufflevector(lo, hi, 0,1,2,3,4,5,6,7,8,9,10,11,12,13,14,15);
}

// B-fragment: pre-packed so each lane reads its 16 halves contiguously (32B).
__device__ __forceinline__ v16h load_bfrag(const _Float16* Bpack, int gntile,
                                           int kc, int lane) {
  const _Float16* p = Bpack + (((size_t)gntile * 4 + kc) * 32 + lane) * 16;
  h8 lo = *(const h8*)(p);
  h8 hi = *(const h8*)(p + 8);
  return __builtin_shufflevector(lo, hi, 0,1,2,3,4,5,6,7,8,9,10,11,12,13,14,15);
}

__device__ __forceinline__ float gelu_f(float x) {
  return 0.5f * x * (1.0f + erff(x * 0.70710678118654752f));
}

// ---------------------------------------------------------------------------
// Prep: packed weight fragments + relbias constants.
// Wpack ntiles: [0,8)=Wc_d [8,16)=Wc_r [16,24)=Wn_d [24,32)=Wn_r
//              [32,40)=vW [40,48)=oW
// ---------------------------------------------------------------------------
__global__ __launch_bounds__(256)
void prep_kernel(const float* __restrict__ dW1, const float* __restrict__ db1,
                 const float* __restrict__ rW1, const float* __restrict__ rb1,
                 const float* __restrict__ vW,  const float* __restrict__ oW,
                 const float* __restrict__ rel_pos,
                 _Float16* __restrict__ Wpack, float* __restrict__ relb) {
  int gid = blockIdx.x * 256 + threadIdx.x;
  if (gid < 48 * 2048) {
    int ntile = gid >> 11;
    int rem   = gid & 2047;
    int kc    = rem >> 9;
    int lane  = (rem >> 4) & 31;
    int e     = rem & 15;
    int k     = kc * 32 + ((lane >> 4) << 3) + (e & 7) + ((e >> 3) << 4);   // K row 0..127
    int ncol  = lane & 15;
    float val;
    if (ntile < 8)       { int h = ntile*16 + ncol;      val = dW1[k*128+h] + dW1[(256+k)*128+h]; }
    else if (ntile < 16) { int h = (ntile-8)*16 + ncol;  val = rW1[k*128+h] + rW1[(256+k)*128+h]; }
    else if (ntile < 24) { int h = (ntile-16)*16 + ncol; val = dW1[(128+k)*128+h] - dW1[(256+k)*128+h]; }
    else if (ntile < 32) { int h = (ntile-24)*16 + ncol; val = rW1[(128+k)*128+h] - rW1[(256+k)*128+h]; }
    else if (ntile < 40) { int h = (ntile-32)*16 + ncol; val = vW[k*128+h]; }
    else                 { int h = (ntile-40)*16 + ncol; val = oW[k*128+h]; }
    Wpack[gid] = (_Float16)val;
  } else if (gid < 48 * 2048 + 2304) {
    int q = gid - 48 * 2048;
    int which = q / 1152;               // 0=drive, 1=resistance
    int r2 = q - which * 1152;
    int n = r2 >> 7;
    int h = r2 & 127;
    const float* W1 = which ? rW1 : dW1;
    const float* b1 = which ? rb1 : db1;
    float acc = b1[h];
#pragma unroll
    for (int rr = 0; rr < 8; ++rr) acc += rel_pos[n*8+rr] * W1[(384+rr)*128 + h];
    relb[q] = acc;
  }
}

// ---------------------------------------------------------------------------
// LayerNorm 1: tokens f32 -> normalized f16 (one wave per token, 4 ch/lane)
// ---------------------------------------------------------------------------
__global__ __launch_bounds__(256)
void ln1_kernel(const float* __restrict__ tokens, const float* __restrict__ g,
                const float* __restrict__ b, _Float16* __restrict__ Nrm) {
  int w = threadIdx.x >> 5, lane = threadIdx.x & 31;
  int l = blockIdx.x * 8 + w;
  int c = lane * 4;
  const float4 xv = *(const float4*)(tokens + (size_t)l * 128 + c);
  float s = xv.x + xv.y + xv.z + xv.w;
#pragma unroll
  for (int m = 1; m < 32; m <<= 1) s += __shfl_xor(s, m, 32);
  float mean = s * (1.0f / 128.0f);
  float d0 = xv.x - mean, d1 = xv.y - mean, d2 = xv.z - mean, d3 = xv.w - mean;
  float ss = d0*d0 + d1*d1 + d2*d2 + d3*d3;
#pragma unroll
  for (int m = 1; m < 32; m <<= 1) ss += __shfl_xor(ss, m, 32);
  float inv = rsqrtf(ss * (1.0f / 128.0f) + LNE);
  h4 o;
  o[0] = (_Float16)(d0 * inv * g[c+0] + b[c+0]);
  o[1] = (_Float16)(d1 * inv * g[c+1] + b[c+1]);
  o[2] = (_Float16)(d2 * inv * g[c+2] + b[c+2]);
  o[3] = (_Float16)(d3 * inv * g[c+3] + b[c+3]);
  *(h4*)(Nrm + (size_t)l * 128 + c) = o;
}

// ---------------------------------------------------------------------------
// Generic WMMA GEMM: C[M x ncols] = A[M x 128] @ Wpack[ntile_base ...]
// ---------------------------------------------------------------------------
template <typename OutT>
__global__ __launch_bounds__(256)
void gemm128(const _Float16* __restrict__ A, const _Float16* __restrict__ Bpack,
             OutT* __restrict__ C, int ncols, int ntile_base) {
  int lane = threadIdx.x & 31;
  int w    = threadIdx.x >> 5;
  int mbase = blockIdx.x * 64;
  int gnt   = blockIdx.y * 8 + w;
  v8f acc[4] = {};
#pragma unroll
  for (int kc = 0; kc < 4; ++kc) {
    v16h bf = load_bfrag(Bpack, ntile_base + gnt, kc, lane);
#pragma unroll
    for (int mt = 0; mt < 4; ++mt) {
      int row = mbase + mt * 16 + (lane & 15);
      v16h af = load_afrag(A, row, 128, kc, lane);
      acc[mt] = wmma_f16(af, bf, acc[mt]);
    }
  }
  int n    = gnt * 16 + (lane & 15);
  int roff = (lane >> 4) * 8;
#pragma unroll
  for (int mt = 0; mt < 4; ++mt)
#pragma unroll
    for (int j = 0; j < 8; ++j) {
      int m = mbase + mt * 16 + roff + j;
      C[(size_t)m * ncols + n] = (OutT)acc[mt][j];
    }
}

// ---------------------------------------------------------------------------
// Mega kernel: 14 tokens/block. Phases are serialized so only ONE set of 8
// WMMA accumulators is live at any time (no spills @ launch_bounds(256)).
// ---------------------------------------------------------------------------
__global__ __launch_bounds__(256)
void mega_kernel(const _Float16* __restrict__ Nrm,
                 const _Float16* __restrict__ Acent,
                 const _Float16* __restrict__ Wpack,
                 const float* __restrict__ relb,
                 const float* __restrict__ dW2, const float* __restrict__ db2,
                 const float* __restrict__ rW2, const float* __restrict__ rb2,
                 const float* __restrict__ vb,
                 const float* __restrict__ gam_p, const float* __restrict__ lam_p,
                 const float* __restrict__ bias_p,
                 _Float16* __restrict__ Msg) {
  __shared__ __align__(16) _Float16 Z[128 * 128];     // Z tile; reused as V tile
  __shared__ __align__(16) _Float16 AcentS[TPB * 256];// staged center terms
  __shared__ float relbS[2 * 9 * 128];
  __shared__ float dW2S[128], rW2S[128], vbS[128];
  __shared__ float gates[128];

  int tid = threadIdx.x, lane = tid & 31, w = tid >> 5;
  int base = blockIdx.x * TPB;

  for (int q = tid; q < 2304; q += 256) relbS[q] = relb[q];
  if (tid < 128) { dW2S[tid] = dW2[tid]; rW2S[tid] = rW2[tid]; vbS[tid] = vb[tid]; }

  // stage Acent rows for this block's tokens (coalesced f16 loads -> LDS)
  for (int q = tid; q < TPB * 256; q += 256) {
    int t = q >> 8, col = q & 255;
    int l = base + t; if (l > BL - 1) l = BL - 1;
    AcentS[q] = Acent[(size_t)l * 256 + col];
  }

  // gather permuted neighborhood: Z[t*9+n][c] = Nrm[nb_s(l)][c0]
  //   s = (n*128+c)%9, c0 = (n*128+c)/9; zero for OOB / pad rows
  for (int q = tid; q < 128 * 128; q += 256) {
    int r = q >> 7, c = q & 127;
    _Float16 val = (_Float16)0.0f;
    int t = r / 9;
    if (r < 126 && base + t < BL) {
      int n = r - t * 9;
      int idx = n * 128 + c;
      int s = idx % 9, c0 = idx / 9;
      int l = base + t;
      int bb = l >> 14; int rem = l & 16383; int y = rem >> 7; int x = rem & 127;
      int ny = y + s / 3 - 1, nx = x + s % 3 - 1;
      if ((unsigned)ny < 128u && (unsigned)nx < 128u) {
        int nl = (bb << 14) + (ny << 7) + nx;
        val = Nrm[(size_t)nl * 128 + c0];
      }
    }
    Z[q] = val;
  }
  __syncthreads();

  int roff = (lane >> 4) * 8;
  int hcol = lane & 15;
  int arow = w * 16 + (lane & 15);

  // ---- phase D: drive hidden -> fold into per-row dot partials, retire accs
  float dotd[8] = {};
  {
    v8f acc[8] = {};
#pragma unroll
    for (int kc = 0; kc < 4; ++kc) {
      v16h a = load_afrag(Z, arow, 128, kc, lane);
      v16h bcur = load_bfrag(Wpack, 16, kc, lane);
#pragma unroll
      for (int nt = 0; nt < 8; ++nt) {
        v16h bnxt = bcur;
        if (nt < 7) bnxt = load_bfrag(Wpack, 16 + nt + 1, kc, lane);
        acc[nt] = wmma_f16(a, bcur, acc[nt]);
        bcur = bnxt;
      }
    }
#pragma unroll
    for (int j = 0; j < 8; ++j) {
      int r = w * 16 + roff + j;
      int t = r / 9; int tc = (t > TPB - 1) ? (TPB - 1) : t;
      int n = r - t * 9;
#pragma unroll
      for (int nt = 0; nt < 8; ++nt) {
        int h = nt * 16 + hcol;
        float hd = acc[nt][j] + (float)AcentS[tc * 256 + h] + relbS[n * 128 + h];
        dotd[j] += gelu_f(hd) * dW2S[h];
      }
    }
  }

  // ---- phase R: resistance hidden -> dot partials
  float dotr[8] = {};
  {
    v8f acc[8] = {};
#pragma unroll
    for (int kc = 0; kc < 4; ++kc) {
      v16h a = load_afrag(Z, arow, 128, kc, lane);
      v16h bcur = load_bfrag(Wpack, 24, kc, lane);
#pragma unroll
      for (int nt = 0; nt < 8; ++nt) {
        v16h bnxt = bcur;
        if (nt < 7) bnxt = load_bfrag(Wpack, 24 + nt + 1, kc, lane);
        acc[nt] = wmma_f16(a, bcur, acc[nt]);
        bcur = bnxt;
      }
    }
#pragma unroll
    for (int j = 0; j < 8; ++j) {
      int r = w * 16 + roff + j;
      int t = r / 9; int tc = (t > TPB - 1) ? (TPB - 1) : t;
      int n = r - t * 9;
#pragma unroll
      for (int nt = 0; nt < 8; ++nt) {
        int h = nt * 16 + hcol;
        float hr = acc[nt][j] + (float)AcentS[tc * 256 + 128 + h] + relbS[1152 + n * 128 + h];
        dotr[j] += gelu_f(hr) * rW2S[h];
      }
    }
  }

  // ---- EML gate per row
  {
    float db2v = db2[0], rb2v = rb2[0];
    float gam = gam_p[0], lam = lam_p[0], bias = bias_p[0];
#pragma unroll
    for (int j = 0; j < 8; ++j) {
      float dd = dotd[j], dr = dotr[j];
#pragma unroll
      for (int m = 1; m < 16; m <<= 1) {       // reduce within 16-lane half
        dd += __shfl_xor(dd, m, 32);
        dr += __shfl_xor(dr, m, 32);
      }
      float drive = dd + db2v;
      float res   = dr + rb2v;
      float sp = (res > 20.0f) ? res : log1pf(expf(res));
      float e = gam * drive / (lam * sp + GEPS) + bias;
      e = fminf(fmaxf(e, -CLIPV), CLIPV);
      float g = 1.0f / (1.0f + expf(-e));
      int r = w * 16 + roff + j;
      if (hcol == 0 && r < 126) gates[r] = g;
    }
  }

  // ---- phase V: values = Z @ vW
  v8f accv[8] = {};
#pragma unroll
  for (int kc = 0; kc < 4; ++kc) {
    v16h a = load_afrag(Z, arow, 128, kc, lane);
    v16h bcur = load_bfrag(Wpack, 32, kc, lane);
#pragma unroll
    for (int nt = 0; nt < 8; ++nt) {
      v16h bnxt = bcur;
      if (nt < 7) bnxt = load_bfrag(Wpack, 32 + nt + 1, kc, lane);
      accv[nt] = wmma_f16(a, bcur, accv[nt]);
      bcur = bnxt;
    }
  }
  __syncthreads();                             // Z reads done; gates visible

  _Float16* V = Z;                             // alias LDS
#pragma unroll
  for (int nt = 0; nt < 8; ++nt)
#pragma unroll
    for (int j = 0; j < 8; ++j)
      V[(w * 16 + roff + j) * 128 + nt * 16 + hcol] = (_Float16)accv[nt][j];
  __syncthreads();

  // ---- gated message: msg = (sum_n g*V + sum_g*vb) / max(sum_g, eps)
  for (int q = tid; q < TPB * 128; q += 256) {
    int t = q >> 7, c = q & 127;
    int l = base + t;
    if (l < BL) {
      float sumg = 0.0f, msum = 0.0f;
#pragma unroll
      for (int n = 0; n < 9; ++n) {
        float g = gates[t * 9 + n];
        sumg += g;
        msum += g * (float)V[(t * 9 + n) * 128 + c];
      }
      float mass = fmaxf(sumg, GEPS);
      Msg[(size_t)l * 128 + c] = (_Float16)((msum + sumg * vbS[c]) / mass);
    }
  }
}

// ---------------------------------------------------------------------------
// LayerNorm 2: out = LN(tokens + update + ob)
// ---------------------------------------------------------------------------
__global__ __launch_bounds__(256)
void ln2_kernel(const float* __restrict__ tokens, const float* __restrict__ upd,
                const float* __restrict__ ob,
                const float* __restrict__ g, const float* __restrict__ b,
                float* __restrict__ out) {
  int w = threadIdx.x >> 5, lane = threadIdx.x & 31;
  int l = blockIdx.x * 8 + w;
  int c = lane * 4;
  const float4 tv = *(const float4*)(tokens + (size_t)l * 128 + c);
  const float4 uv = *(const float4*)(upd    + (size_t)l * 128 + c);
  float x0 = tv.x + uv.x + ob[c+0];
  float x1 = tv.y + uv.y + ob[c+1];
  float x2 = tv.z + uv.z + ob[c+2];
  float x3 = tv.w + uv.w + ob[c+3];
  float s = x0 + x1 + x2 + x3;
#pragma unroll
  for (int m = 1; m < 32; m <<= 1) s += __shfl_xor(s, m, 32);
  float mean = s * (1.0f / 128.0f);
  float d0 = x0 - mean, d1 = x1 - mean, d2 = x2 - mean, d3 = x3 - mean;
  float ss = d0*d0 + d1*d1 + d2*d2 + d3*d3;
#pragma unroll
  for (int m = 1; m < 32; m <<= 1) ss += __shfl_xor(ss, m, 32);
  float inv = rsqrtf(ss * (1.0f / 128.0f) + LNE);
  float4 o;
  o.x = d0 * inv * g[c+0] + b[c+0];
  o.y = d1 * inv * g[c+1] + b[c+1];
  o.z = d2 * inv * g[c+2] + b[c+2];
  o.w = d3 * inv * g[c+3] + b[c+3];
  *(float4*)(out + (size_t)l * 128 + c) = o;
}

// ---------------------------------------------------------------------------
extern "C" void kernel_launch(void* const* d_in, const int* in_sizes, int n_in,
                              void* d_out, int out_size, void* d_ws, size_t ws_size,
                              hipStream_t stream) {
  const float* tokens = (const float*)d_in[0];
  const float* ln1_g  = (const float*)d_in[1];
  const float* ln1_b  = (const float*)d_in[2];
  const float* ln2_g  = (const float*)d_in[3];
  const float* ln2_b  = (const float*)d_in[4];
  const float* rel_pos= (const float*)d_in[5];
  const float* dW1    = (const float*)d_in[6];
  const float* db1    = (const float*)d_in[7];
  const float* dW2    = (const float*)d_in[8];
  const float* db2    = (const float*)d_in[9];
  const float* rW1    = (const float*)d_in[10];
  const float* rb1    = (const float*)d_in[11];
  const float* rW2    = (const float*)d_in[12];
  const float* rb2    = (const float*)d_in[13];
  const float* vW     = (const float*)d_in[14];
  const float* vb     = (const float*)d_in[15];
  const float* oW     = (const float*)d_in[16];
  const float* ob     = (const float*)d_in[17];
  const float* gam    = (const float*)d_in[18];
  const float* lam    = (const float*)d_in[19];
  const float* bias   = (const float*)d_in[20];

  char* ws = (char*)d_ws;
  _Float16* Nrm   = (_Float16*)(ws + 0);           //  8 MB: [BL,128] f16
  _Float16* Acent = (_Float16*)(ws + 8388608);     // 16 MB: [BL,256] f16
  _Float16* Msg   = (_Float16*)(ws + 25165824);    //  8 MB: [BL,128] f16
  float*    Upd   = (float*)   (ws + 33554432);    // 16 MB: [BL,128] f32
  _Float16* Wpack = (_Float16*)(ws + 50331648);    // 192 KB packed fragments
  float*    relb  = (float*)   (ws + 50528256);    // 9 KB  relbias

  prep_kernel<<<393, 256, 0, stream>>>(dW1, db1, rW1, rb1, vW, oW, rel_pos, Wpack, relb);
  ln1_kernel<<<BL / 8, 256, 0, stream>>>(tokens, ln1_g, ln1_b, Nrm);
  gemm128<_Float16><<<dim3(BL / 64, 2), 256, 0, stream>>>(Nrm, Wpack, Acent, 256, 0);
  mega_kernel<<<(BL + TPB - 1) / TPB, 256, 0, stream>>>(Nrm, Acent, Wpack, relb,
                                                        dW2, db2, rW2, rb2, vb,
                                                        gam, lam, bias, Msg);
  gemm128<float><<<dim3(BL / 64, 1), 256, 0, stream>>>(Msg, Wpack, Upd, 128, 40);
  ln2_kernel<<<BL / 8, 256, 0, stream>>>(tokens, Upd, ob, ln2_g, ln2_b, (float*)d_out);
}
